// MAScore_23613730194182
// MI455X (gfx1250) — compile-verified, matrix operand order checked by reading, not run on hardware
//
#include <hip/hip_runtime.h>
#include <hip/hip_bf16.h>
#include <math.h>

// ---------------------------------------------------------------- types
typedef _Float16 half_t;
typedef __attribute__((ext_vector_type(16))) _Float16 v16h;
typedef __attribute__((ext_vector_type(8)))  _Float16 v8h;
typedef __attribute__((ext_vector_type(8)))  float    v8f;

union Frag { v16h v; v8h h[2]; };

// ---------------------------------------------------------------- sizes
constexpr int BB   = 8;
constexpr int C    = 256;
constexpr int NN   = 2048;
constexpr int H    = 4;
constexpr int KCLS = 64;
constexpr int HD   = 64;     // C / H
constexpr int M    = BB * NN; // 16384 total rows

constexpr size_t MBY = 1ull << 20;
// workspace layout (bytes), with lifetime-based reuse:
constexpr size_t OFF_XT  = 0;        // 8MB  f16 [M,C]   (later: attn out o_h, later: h2_h)
constexpr size_t OFF_QH  = 8*MBY;    // 8MB  f16 [B,H,N,HD] (later: h1_h, later: logits f32 4MB)
constexpr size_t OFF_KH  = 16*MBY;   // 8MB  f16 [B,H,N,HD] (later: ff1 act f16 16MB spans KH+VT)
constexpr size_t OFF_VT  = 24*MBY;   // 8MB  f16 [B,H,HD,N]
constexpr size_t OFF_OPJ = 32*MBY;   // 16MB f32 [M,C]    (later: ff2 out f32)
constexpr size_t OFF_H1  = 48*MBY;   // 16MB f32 [M,C]
constexpr size_t OFF_W   = 64*MBY;   // f16 weights:
constexpr size_t OFF_WCQKV = OFF_W;                 // 3*C*C f16 = 384KB
constexpr size_t OFF_WOUT  = OFF_W +  384*1024;     // C*C   f16 = 128KB
constexpr size_t OFF_WFF1  = OFF_W +  512*1024;     // 2C*C  f16 = 256KB
constexpr size_t OFF_WFF2  = OFF_W +  768*1024;     // C*2C  f16 = 256KB
constexpr size_t OFF_WPROJ = OFF_W + 1024*1024;     // K*C   f16 = 32KB

// ---------------------------------------------------------------- helpers
__device__ __forceinline__ float gelu_exact(float x) {
    return 0.5f * x * (1.0f + erff(x * 0.70710678118654752f));
}

// ---------------------------------------------------------------- x [B,C,N] f32 -> xt [B,N,C] f16
__global__ __launch_bounds__(256) void transpose_x_kernel(const float* __restrict__ x,
                                                          half_t* __restrict__ xt) {
    __shared__ float tile[32][33];
    const int b  = blockIdx.z;
    const int n0 = blockIdx.x * 32, c0 = blockIdx.y * 32;
    const int tx = threadIdx.x, ty = threadIdx.y;   // block (32,8)
    #pragma unroll
    for (int yy = ty; yy < 32; yy += 8)
        tile[yy][tx] = x[((size_t)b * C + c0 + yy) * NN + n0 + tx];
    __syncthreads();
    #pragma unroll
    for (int yy = ty; yy < 32; yy += 8)
        xt[((size_t)b * NN + n0 + yy) * C + c0 + tx] = (half_t)tile[tx][yy];
}

// ---------------------------------------------------------------- fold WQ/WK/WV into MHA in_proj
// Wc[mat][o][i] = sum_j W1[i][j] * in_proj_w[mat*C + o][j]   (f16, GEMM-B layout: rows along i=k)
__global__ __launch_bounds__(256) void combine_w_kernel(const float* __restrict__ WQ,
                                                        const float* __restrict__ WK,
                                                        const float* __restrict__ WV,
                                                        const float* __restrict__ ipw,
                                                        half_t* __restrict__ out) {
    const int idx = blockIdx.x * 256 + threadIdx.x;   // 3*C*C threads
    const int mat = idx / (C * C);
    const int rem = idx % (C * C);
    const int o = rem / C, i = rem % C;
    const float* W1 = (mat == 0) ? WQ : (mat == 1 ? WK : WV);
    const float* wrow = ipw + (size_t)(mat * C + o) * C;
    const float* xrow = W1 + (size_t)i * C;
    float acc = 0.0f;
    #pragma unroll 8
    for (int j = 0; j < C; ++j) acc += xrow[j] * wrow[j];
    out[idx] = (half_t)acc;
}

__global__ __launch_bounds__(256) void cvt_f32_f16_kernel(const float* __restrict__ s,
                                                          half_t* __restrict__ d, int n) {
    const int i = blockIdx.x * 256 + threadIdx.x;
    if (i < n) d[i] = (half_t)s[i];
}

// ---------------------------------------------------------------- fused QKV GEMM (WMMA)
// [M,C] x Wc^T + b ; z=0 -> qh[B,H,N,HD], z=1 -> kh[B,H,N,HD], z=2 -> vt[B,H,HD,N]
__global__ __launch_bounds__(128) void qkv_gemm_kernel(const half_t* __restrict__ xt,
                                                       const half_t* __restrict__ wc,
                                                       const float* __restrict__ ipb,
                                                       half_t* __restrict__ qh,
                                                       half_t* __restrict__ kh,
                                                       half_t* __restrict__ vt) {
    const int lane = threadIdx.x & 31, wave = threadIdx.x >> 5;
    const int r = lane & 15, sel = lane >> 4;
    const int z  = blockIdx.z;
    const int m0 = (blockIdx.x * 4 + wave) * 16;
    const int n0 = blockIdx.y * 64;
    const half_t* W    = wc + (size_t)z * C * C;
    const float*  bias = ipb + z * C;

    v8f acc[4];
    #pragma unroll
    for (int j = 0; j < 4; ++j) {
        const float bz = bias[n0 + 16 * j + r];
        #pragma unroll
        for (int i = 0; i < 8; ++i) acc[j][i] = bz;
    }
    const half_t* arow = xt + (size_t)(m0 + r) * C;
    #pragma unroll 2
    for (int kb = 0; kb < C; kb += 32) {
        Frag af;
        af.h[0] = *(const v8h*)(arow + kb + sel * 8);
        af.h[1] = *(const v8h*)(arow + kb + 16 + sel * 8);
        #pragma unroll
        for (int j = 0; j < 4; ++j) {
            const half_t* wr = W + (size_t)(n0 + 16 * j + r) * C + kb;
            Frag bf;
            bf.h[0] = *(const v8h*)(wr + sel * 8);
            bf.h[1] = *(const v8h*)(wr + 16 + sel * 8);
            acc[j] = __builtin_amdgcn_wmma_f32_16x16x32_f16(false, af.v, false, bf.v,
                                                            (short)0, acc[j], false, false);
        }
    }
    #pragma unroll
    for (int j = 0; j < 4; ++j) {
        const int n = n0 + 16 * j + r;            // channel
        const int hh = n >> 6, d = n & 63;
        #pragma unroll
        for (int i = 0; i < 8; ++i) {
            const int m = m0 + sel * 8 + i;       // b*N + nseq
            const int b = m >> 11, nseq = m & (NN - 1);
            const half_t hv = (half_t)acc[j][i];
            if (z == 0)      qh[(((size_t)b * H + hh) * NN + nseq) * HD + d] = hv;
            else if (z == 1) kh[(((size_t)b * H + hh) * NN + nseq) * HD + d] = hv;
            else             vt[(((size_t)b * H + hh) * HD + d) * NN + nseq] = hv;
        }
    }
}

// ---------------------------------------------------------------- flash attention (WMMA, S^T path)
__global__ __launch_bounds__(128) void attn_kernel(const half_t* __restrict__ qh,
                                                   const half_t* __restrict__ kh,
                                                   const half_t* __restrict__ vt,
                                                   const int* __restrict__ lengths,
                                                   half_t* __restrict__ o_out) {
    const int lane = threadIdx.x & 31, wave = threadIdx.x >> 5;
    const int r = lane & 15, sel = lane >> 4;
    const int h = blockIdx.y, b = blockIdx.z;
    const int q0 = (blockIdx.x * 4 + wave) * 16;
    const int len = lengths[b];
    const float scale = 0.125f;   // 1/sqrt(HD)

    // Q fragments (B-operand of S^T = K * Q^T): lane -> query r, elements -> d
    const half_t* qbase = qh + (((size_t)b * H + h) * NN + q0 + r) * HD;
    Frag qf[2];
    #pragma unroll
    for (int c = 0; c < 2; ++c) {
        qf[c].h[0] = *(const v8h*)(qbase + 32 * c + sel * 8);
        qf[c].h[1] = *(const v8h*)(qbase + 32 * c + 16 + sel * 8);
    }

    float mrun = -3.0e38f, lrun = 0.0f;
    v8f oacc[4];
    #pragma unroll
    for (int t = 0; t < 4; ++t)
        #pragma unroll
        for (int i = 0; i < 8; ++i) oacc[t][i] = 0.0f;

    const half_t* krow = kh + ((size_t)b * H + h) * NN * HD;
    const half_t* vrow = vt + ((size_t)b * H + h) * HD * NN;
    const int kend = (len + 31) & ~31;

    for (int kb = 0; kb < kend; kb += 32) {
        v8f s[2];
        #pragma unroll
        for (int t = 0; t < 2; ++t) {
            const half_t* kp = krow + (size_t)(kb + 16 * t + r) * HD;
            Frag kf0, kf1;
            kf0.h[0] = *(const v8h*)(kp + sel * 8);
            kf0.h[1] = *(const v8h*)(kp + 16 + sel * 8);
            kf1.h[0] = *(const v8h*)(kp + 32 + sel * 8);
            kf1.h[1] = *(const v8h*)(kp + 48 + sel * 8);
            v8f z;
            #pragma unroll
            for (int i = 0; i < 8; ++i) z[i] = 0.0f;
            z = __builtin_amdgcn_wmma_f32_16x16x32_f16(false, kf0.v, false, qf[0].v,
                                                       (short)0, z, false, false);
            z = __builtin_amdgcn_wmma_f32_16x16x32_f16(false, kf1.v, false, qf[1].v,
                                                       (short)0, z, false, false);
            s[t] = z;
        }
        // scale + key mask + online softmax (per-q state lives in lanes r and r+16)
        float p[16];
        float bm = -3.0e38f;
        #pragma unroll
        for (int t = 0; t < 2; ++t)
            #pragma unroll
            for (int i = 0; i < 8; ++i) {
                const int kk = kb + 16 * t + sel * 8 + i;
                const float v = (kk < len) ? s[t][i] * scale : -1.0e30f;
                p[t * 8 + i] = v;
                bm = fmaxf(bm, v);
            }
        bm = fmaxf(bm, __shfl_xor(bm, 16, 32));
        const float mnew = fmaxf(mrun, bm);
        const float sc = __expf(mrun - mnew);
        mrun = mnew;
        lrun *= sc;
        Frag pf;   // P^T B-fragment: elements 0..7 = key tile0 rows, 8..15 = tile1 rows
        #pragma unroll
        for (int i = 0; i < 16; ++i) {
            const float e = __expf(p[i] - mnew);
            lrun += e;
            pf.v[i] = (half_t)e;
        }
        #pragma unroll
        for (int t = 0; t < 4; ++t)
            #pragma unroll
            for (int i = 0; i < 8; ++i) oacc[t][i] *= sc;
        #pragma unroll
        for (int t = 0; t < 4; ++t) {
            const half_t* vp = vrow + (size_t)(16 * t + r) * NN + kb;
            Frag vf;
            vf.h[0] = *(const v8h*)(vp + sel * 8);
            vf.h[1] = *(const v8h*)(vp + 16 + sel * 8);
            oacc[t] = __builtin_amdgcn_wmma_f32_16x16x32_f16(false, vf.v, false, pf.v,
                                                             (short)0, oacc[t], false, false);
        }
    }
    const float ltot = lrun + __shfl_xor(lrun, 16, 32);
    const float inv = 1.0f / ltot;
    // O^T D-layout: lane column = q (= r), rows = d = 16t + sel*8 + i  -> contiguous stores
    half_t* orow = o_out + ((size_t)b * NN + q0 + r) * C + h * HD;
    #pragma unroll
    for (int t = 0; t < 4; ++t) {
        v8h pk;
        #pragma unroll
        for (int i = 0; i < 8; ++i) pk[i] = (half_t)(oacc[t][i] * inv);
        *(v8h*)(orow + 16 * t + sel * 8) = pk;
    }
}

// ---------------------------------------------------------------- generic WMMA GEMM: out = A @ W^T + b
// EPI: 0 = f16 out, 1 = f32 out, 2 = gelu -> f16 out
template <int EPI>
__global__ __launch_bounds__(128) void gemm_kernel(const half_t* __restrict__ A,
                                                   const half_t* __restrict__ W,
                                                   const float* __restrict__ bias,
                                                   void* __restrict__ outp,
                                                   int Kd, int Nc) {
    const int lane = threadIdx.x & 31, wave = threadIdx.x >> 5;
    const int r = lane & 15, sel = lane >> 4;
    const int m0 = (blockIdx.x * 4 + wave) * 16;
    const int n0 = blockIdx.y * 64;

    v8f acc[4];
    #pragma unroll
    for (int j = 0; j < 4; ++j) {
        const float bz = bias ? bias[n0 + 16 * j + r] : 0.0f;
        #pragma unroll
        for (int i = 0; i < 8; ++i) acc[j][i] = bz;
    }
    const half_t* arow = A + (size_t)(m0 + r) * Kd;
    for (int kb = 0; kb < Kd; kb += 32) {
        Frag af;
        af.h[0] = *(const v8h*)(arow + kb + sel * 8);
        af.h[1] = *(const v8h*)(arow + kb + 16 + sel * 8);
        #pragma unroll
        for (int j = 0; j < 4; ++j) {
            const half_t* wr = W + (size_t)(n0 + 16 * j + r) * Kd + kb;
            Frag bf;
            bf.h[0] = *(const v8h*)(wr + sel * 8);
            bf.h[1] = *(const v8h*)(wr + 16 + sel * 8);
            acc[j] = __builtin_amdgcn_wmma_f32_16x16x32_f16(false, af.v, false, bf.v,
                                                            (short)0, acc[j], false, false);
        }
    }
    #pragma unroll
    for (int j = 0; j < 4; ++j) {
        const int n = n0 + 16 * j + r;
        #pragma unroll
        for (int i = 0; i < 8; ++i) {
            const int m = m0 + sel * 8 + i;
            float v = acc[j][i];
            if (EPI == 2) v = gelu_exact(v);
            if (EPI == 1) ((float*)outp)[(size_t)m * Nc + n] = v;
            else          ((half_t*)outp)[(size_t)m * Nc + n] = (half_t)v;
        }
    }
}

// ---------------------------------------------------------------- LayerNorm over C, residual add
// a_is_bcn: src_a is original x [B,C,N]; else [M,C] f32. out32 may be null.
__global__ __launch_bounds__(256) void ln_kernel(const float* __restrict__ src_a, int a_is_bcn,
                                                 const float* __restrict__ src_b,
                                                 const float* __restrict__ w,
                                                 const float* __restrict__ bia,
                                                 float* __restrict__ out32,
                                                 half_t* __restrict__ out16) {
    const int lane = threadIdx.x & 31, wave = threadIdx.x >> 5;
    const int row = blockIdx.x * 8 + wave;
    const int b = row >> 11, n = row & (NN - 1);
    float v[8], s = 0.0f, s2 = 0.0f;
    #pragma unroll
    for (int i = 0; i < 8; ++i) {
        const int c = lane + 32 * i;
        const float va = a_is_bcn ? src_a[((size_t)b * C + c) * NN + n]
                                  : src_a[(size_t)row * C + c];
        const float t = va + src_b[(size_t)row * C + c];
        v[i] = t; s += t; s2 += t * t;
    }
    #pragma unroll
    for (int off = 16; off >= 1; off >>= 1) {
        s  += __shfl_xor(s, off, 32);
        s2 += __shfl_xor(s2, off, 32);
    }
    const float mean = s * (1.0f / C);
    const float var  = s2 * (1.0f / C) - mean * mean;
    const float rstd = rsqrtf(var + 1e-5f);
    #pragma unroll
    for (int i = 0; i < 8; ++i) {
        const int c = lane + 32 * i;
        const float y = (v[i] - mean) * rstd * w[c] + bia[c];
        if (out32) out32[(size_t)row * C + c] = y;
        out16[(size_t)row * C + c] = (half_t)y;
    }
}

// ---------------------------------------------------------------- masked softmax over sequence axis
// logits [B,N,K] f32 -> out [B,K,N] f32 (zero at invalid positions)
__global__ __launch_bounds__(256) void seqsm_kernel(const float* __restrict__ logits,
                                                    const int* __restrict__ lengths,
                                                    float* __restrict__ out) {
    __shared__ float red[8];
    const int bk = blockIdx.x;
    const int b = bk >> 6, kc = bk & (KCLS - 1);
    const int len = lengths[b];
    const int tid = threadIdx.x, lane = tid & 31, wave = tid >> 5;
    float v[8], mx = -3.0e38f;
    #pragma unroll
    for (int i = 0; i < 8; ++i) {
        const int n = tid + i * 256;
        const float t = (n < len) ? logits[((size_t)b * NN + n) * KCLS + kc] : -3.0e38f;
        v[i] = t; mx = fmaxf(mx, t);
    }
    #pragma unroll
    for (int off = 16; off >= 1; off >>= 1) mx = fmaxf(mx, __shfl_xor(mx, off, 32));
    if (lane == 0) red[wave] = mx;
    __syncthreads();
    float bmx = red[0];
    #pragma unroll
    for (int wv = 1; wv < 8; ++wv) bmx = fmaxf(bmx, red[wv]);
    __syncthreads();
    float e[8], sum = 0.0f;
    #pragma unroll
    for (int i = 0; i < 8; ++i) {
        const int n = tid + i * 256;
        const float t = (n < len) ? __expf(v[i] - bmx) : 0.0f;
        e[i] = t; sum += t;
    }
    #pragma unroll
    for (int off = 16; off >= 1; off >>= 1) sum += __shfl_xor(sum, off, 32);
    if (lane == 0) red[wave] = sum;
    __syncthreads();
    float tot = 0.0f;
    #pragma unroll
    for (int wv = 0; wv < 8; ++wv) tot += red[wv];
    const float inv = 1.0f / tot;
    #pragma unroll
    for (int i = 0; i < 8; ++i) {
        const int n = tid + i * 256;
        out[((size_t)b * KCLS + kc) * NN + n] = e[i] * inv;
    }
}

// ---------------------------------------------------------------- launch
extern "C" void kernel_launch(void* const* d_in, const int* in_sizes, int n_in,
                              void* d_out, int out_size, void* d_ws, size_t ws_size,
                              hipStream_t stream) {
    (void)in_sizes; (void)n_in; (void)out_size; (void)ws_size;
    const float* x        = (const float*)d_in[0];
    const int*   lengths  = (const int*)  d_in[1];
    const float* WQ       = (const float*)d_in[2];
    const float* WK       = (const float*)d_in[3];
    const float* WV       = (const float*)d_in[4];
    const float* ipw      = (const float*)d_in[5];
    const float* ipb      = (const float*)d_in[6];
    const float* opw      = (const float*)d_in[7];
    const float* opb      = (const float*)d_in[8];
    const float* ln1w     = (const float*)d_in[9];
    const float* ln1b     = (const float*)d_in[10];
    const float* ln2w     = (const float*)d_in[11];
    const float* ln2b     = (const float*)d_in[12];
    const float* ff1w     = (const float*)d_in[13];
    const float* ff1b     = (const float*)d_in[14];
    const float* ff2w     = (const float*)d_in[15];
    const float* ff2b     = (const float*)d_in[16];
    const float* pw       = (const float*)d_in[17];
    const float* pb       = (const float*)d_in[18];
    float* out = (float*)d_out;

    char* ws = (char*)d_ws;
    half_t* xt_h   = (half_t*)(ws + OFF_XT);
    half_t* qh     = (half_t*)(ws + OFF_QH);
    half_t* kh     = (half_t*)(ws + OFF_KH);
    half_t* vt     = (half_t*)(ws + OFF_VT);
    float*  oproj  = (float*) (ws + OFF_OPJ);
    float*  h1     = (float*) (ws + OFF_H1);
    half_t* o_h    = (half_t*)(ws + OFF_XT);   // reuse xt_h
    half_t* h1_h   = (half_t*)(ws + OFF_QH);   // reuse qh
    half_t* ff1g   = (half_t*)(ws + OFF_KH);   // reuse kh+vt (16MB)
    float*  ff2o   = (float*) (ws + OFF_OPJ);  // reuse oproj
    half_t* h2_h   = (half_t*)(ws + OFF_XT);   // reuse o_h
    float*  logits = (float*) (ws + OFF_QH);   // reuse h1_h
    half_t* wcqkv  = (half_t*)(ws + OFF_WCQKV);
    half_t* wout   = (half_t*)(ws + OFF_WOUT);
    half_t* wff1   = (half_t*)(ws + OFF_WFF1);
    half_t* wff2   = (half_t*)(ws + OFF_WFF2);
    half_t* wproj  = (half_t*)(ws + OFF_WPROJ);

    // 1) transpose + f16 convert x
    transpose_x_kernel<<<dim3(NN / 32, C / 32, BB), dim3(32, 8), 0, stream>>>(x, xt_h);
    // 2) fold projection pairs; convert remaining weights
    combine_w_kernel<<<3 * C * C / 256, 256, 0, stream>>>(WQ, WK, WV, ipw, wcqkv);
    cvt_f32_f16_kernel<<<(C * C) / 256, 256, 0, stream>>>(opw, wout, C * C);
    cvt_f32_f16_kernel<<<(2 * C * C) / 256, 256, 0, stream>>>(ff1w, wff1, 2 * C * C);
    cvt_f32_f16_kernel<<<(2 * C * C) / 256, 256, 0, stream>>>(ff2w, wff2, 2 * C * C);
    cvt_f32_f16_kernel<<<(KCLS * C) / 256, 256, 0, stream>>>(pw, wproj, KCLS * C);
    // 3) QKV projections (single fused pass, head-layout epilogues)
    qkv_gemm_kernel<<<dim3(M / 64, C / 64, 3), 128, 0, stream>>>(xt_h, wcqkv, ipb, qh, kh, vt);
    // 4) flash attention
    attn_kernel<<<dim3(NN / 64, H, BB), 128, 0, stream>>>(qh, kh, vt, lengths, o_h);
    // 5) out projection (f32 out)
    gemm_kernel<1><<<dim3(M / 64, C / 64), 128, 0, stream>>>(o_h, wout, opb, oproj, C, C);
    // 6) LN1: h1 = LN(xt + o)
    ln_kernel<<<M / 8, 256, 0, stream>>>(x, 1 /*bcn*/, oproj, ln1w, ln1b, h1, h1_h);
    // 7) FF1 + GELU (f16 out)
    gemm_kernel<2><<<dim3(M / 64, 2 * C / 64), 128, 0, stream>>>(h1_h, wff1, ff1b, ff1g, C, 2 * C);
    // 8) FF2 (f32 out)
    gemm_kernel<1><<<dim3(M / 64, C / 64), 128, 0, stream>>>(ff1g, wff2, ff2b, ff2o, 2 * C, C);
    // 9) LN2: h2 = LN(h1 + ff)  (f16 only)
    ln_kernel<<<M / 8, 256, 0, stream>>>(h1, 0, ff2o, ln2w, ln2b, nullptr, h2_h);
    // 10) class projection (f32 logits)
    gemm_kernel<1><<<dim3(M / 64, KCLS / 64), 128, 0, stream>>>(h2_h, wproj, pb, logits, C, KCLS);
    // 11) masked sequence softmax -> [B,K,N]
    seqsm_kernel<<<BB * KCLS, 256, 0, stream>>>(logits, lengths, out);
}
